// RelLearnableMultiHeadAttn_48996986912858
// MI455X (gfx1250) — matrix operand order repaired and checked by reading, not compiled, not run on hardware
//
#include <hip/hip_runtime.h>
#include <hip/hip_bf16.h>

// ---------------------------------------------------------------------------
// Transformer-XL RelLearnableMultiHeadAttn for gfx1250 (MI455X).
// bf16 WMMA (v_wmma_f32_16x16x32_bf16) everywhere, fp32 accumulate.
// Flash-attention tiling; rel_shift handled as a banded Q.R^T + shifted gather.
// All fp32 operands are pre-converted to bf16 ONCE so GEMM inner loops are
// pure global_load_b128 -> v_wmma (no per-iteration convert VALU).
// ---------------------------------------------------------------------------

typedef __attribute__((ext_vector_type(16))) __bf16 v16bf;
typedef __attribute__((ext_vector_type(8)))  float  v8f;

#define QLEN 2048
#define BSZ 2
#define NH 12
#define DH 64
#define DM 768
#define ATT_SCALE 0.125f
#define NEG_INF_F (-1e30f)

struct __align__(16) U16B { unsigned int w[4]; };
union FragU { v16bf v; U16B h[2]; unsigned short u[16]; __bf16 b[16]; };

// hardware RNE f32 -> bf16 (lowers to v_cvt_*bf16* on gfx1250)
__device__ __forceinline__ unsigned short f2bf(float x) {
  __bf16 h = (__bf16)x;
  return __builtin_bit_cast(unsigned short, h);
}

__device__ __forceinline__ v8f v8f_zero() {
  v8f c;
#pragma unroll
  for (int i = 0; i < 8; ++i) c[i] = 0.0f;
  return c;
}

__device__ __forceinline__ v8f wmma_bf16(v16bf a, v16bf b, v8f c) {
  return __builtin_amdgcn_wmma_f32_16x16x32_bf16(false, a, false, b, (short)0, c,
                                                 false, false);
}

// ---- A fragment (16x32): lane holds row M=lane%16; VGPR0..3 hold K = hsel*8 + 0..7,
// ---- VGPR4..7 hold K = 16 + hsel*8 + 0..7, hsel = lane/16.  (ISA 7.12.2)
__device__ __forceinline__ v16bf load_a_bf16(const unsigned short* base, int ld,
                                             int row0, int k0, int lane) {
  const unsigned short* p =
      base + (size_t)(row0 + (lane & 15)) * ld + k0 + ((lane >> 4) & 1) * 8;
  FragU f;
  f.h[0] = *(const U16B*)(p);
  f.h[1] = *(const U16B*)(p + 16);
  return f.v;
}

// ---- B fragment (32x16): lane holds column N=n0+lane%16 with K = k0 + (lane/16)*8 + 0..7
// ---- and k0 + 16 + (lane/16)*8 + 0..7 (matches A).  Source layout: src[n*ld + k].
__device__ __forceinline__ v16bf load_b_bf16(const unsigned short* base, int ld,
                                             int n0, int k0, int lane) {
  const unsigned short* p =
      base + (size_t)(n0 + (lane & 15)) * ld + k0 + ((lane >> 4) & 1) * 8;
  FragU f;
  f.h[0] = *(const U16B*)(p);
  f.h[1] = *(const U16B*)(p + 16);
  return f.v;
}

// B fragment with per-lane row guard (out-of-range rows -> zeros)
__device__ __forceinline__ v16bf load_b_bf16_guard(const unsigned short* base, int ld,
                                                   int n0, int k0, int lane, int nmax) {
  const int nidx = n0 + (lane & 15);
  FragU f;
  if (nidx < nmax) {
    const unsigned short* p =
        base + (size_t)nidx * ld + k0 + ((lane >> 4) & 1) * 8;
    f.h[0] = *(const U16B*)(p);
    f.h[1] = *(const U16B*)(p + 16);
  } else {
#pragma unroll
    for (int e = 0; e < 16; ++e) f.u[e] = 0;
  }
  return f.v;
}

// ===========================================================================
// Kernel 0: one-time fp32 -> bf16 conversion (vectorized x4, grid-stride-free).
// ===========================================================================
__global__ __launch_bounds__(256) void cvt_kernel(const float* __restrict__ src,
                                                  unsigned short* __restrict__ dst,
                                                  int n4) {
  const int idx = blockIdx.x * blockDim.x + threadIdx.x;
  if (idx >= n4) return;
  const float4 v = ((const float4*)src)[idx];
  uint2 o;
  o.x = (unsigned)f2bf(v.x) | ((unsigned)f2bf(v.y) << 16);
  o.y = (unsigned)f2bf(v.z) | ((unsigned)f2bf(v.w) << 16);
  ((uint2*)dst)[idx] = o;
}

// ===========================================================================
// Kernel 1: QKV projection.  X(4096x768) * qkv_w^T(768x2304), both bf16.
// Writes bf16: q[b,n,i,d], q+r_w_bias[b,n,i,d], k[b,n,j,d], v^T[b,n,d,j].
// Each wave: one 16x64 output tile, K looped in steps of 32. 96 WMMA/wave.
// ===========================================================================
__global__ __launch_bounds__(256) void qkv_kernel(
    const unsigned short* __restrict__ w_bf, const unsigned short* __restrict__ qkvw_bf,
    const float* __restrict__ r_w_bias,
    unsigned short* __restrict__ q_bf, unsigned short* __restrict__ qrw_bf,
    unsigned short* __restrict__ k_bf, unsigned short* __restrict__ vt_bf) {
  const int lane = threadIdx.x & 31;
  const int wave = blockIdx.x * (blockDim.x >> 5) + (threadIdx.x >> 5);
  const int NCT  = (3 * DM) / 64;              // 36 column tiles
  const int rs   = wave / NCT;                 // 0..255
  const int ct   = wave % NCT;
  const int row0 = rs * 16;
  const int col0 = ct * 64;

  v8f acc[4];
#pragma unroll
  for (int f = 0; f < 4; ++f) acc[f] = v8f_zero();

  for (int k0 = 0; k0 < DM; k0 += 32) {
    v16bf a = load_a_bf16(w_bf, DM, row0, k0, lane);
#pragma unroll
    for (int f = 0; f < 4; ++f) {
      v16bf b = load_b_bf16(qkvw_bf, DM, col0 + f * 16, k0, lane);
      acc[f] = wmma_bf16(a, b, acc[f]);
    }
  }

  const int sec = col0 / DM;                   // 0=Q 1=K 2=V (tiles align w/ heads)
  const int n   = (col0 % DM) / DH;
  const int Mh  = ((lane >> 4) & 1) * 8;
#pragma unroll
  for (int f = 0; f < 4; ++f) {
#pragma unroll
    for (int v = 0; v < 8; ++v) {
      const int M = v + Mh;
      const int grow = row0 + M;
      const int i = grow >> 1, b = grow & 1;   // X rows interleave (i, b)
      const int d = f * 16 + (lane & 15);
      const float val = acc[f][v];
      const size_t head = (size_t)b * NH + n;
      if (sec == 0) {
        const size_t idx = (head * QLEN + i) * DH + d;
        q_bf[idx]   = f2bf(val);
        qrw_bf[idx] = f2bf(val + r_w_bias[n * DH + d]);
      } else if (sec == 1) {
        k_bf[(head * QLEN + i) * DH + d] = f2bf(val);
      } else {
        vt_bf[(head * DH + d) * QLEN + i] = f2bf(val);   // V transposed
      }
    }
  }
}

// ===========================================================================
// Kernel 2: flash attention.  Block = 4 waves, each wave owns a 16-row strip
// of one (b, n, i-tile64).  Causal loop over 64-wide j tiles.
// rel_shift:  BD[i,j] = q_i . r_emb[2047+j-i] + r_bias[2047+j-i]  (j <= i).
// Per strip/j-tile the r index is a 79-wide band -> banded T=Q.R^T in LDS,
// gathered with local column c = jj - ii + 15.
// ===========================================================================
__global__ __launch_bounds__(128) void attn_kernel(
    const unsigned short* __restrict__ q_bf, const unsigned short* __restrict__ qrw_bf,
    const unsigned short* __restrict__ k_bf, const unsigned short* __restrict__ vt_bf,
    const unsigned short* __restrict__ remb_bf, const float* __restrict__ r_bias,
    unsigned short* __restrict__ av_bf) {
  __shared__ __align__(32) float          T_lds[4][16][80];
  __shared__ __align__(32) unsigned short P_lds[4][16][64];

  const int lane = threadIdx.x & 31;
  const int wv   = threadIdx.x >> 5;           // 0..3
  const int it   = blockIdx.x & 31;            // i-tile (64 rows)
  const int bn   = blockIdx.x >> 5;            // b*NH + n
  const int n    = bn % NH;
  const int b    = bn / NH;
  const int i_base = it * 64 + wv * 16;

  const unsigned short* Qp  = q_bf   + (size_t)bn * QLEN * DH;
  const unsigned short* Qrp = qrw_bf + (size_t)bn * QLEN * DH;
  const unsigned short* Kp  = k_bf   + (size_t)bn * QLEN * DH;
  const unsigned short* Vtp = vt_bf  + (size_t)bn * DH * QLEN;
  const unsigned short* Rp  = remb_bf + n * DH;            // rows m, ld = DM

  v16bf aq[2], aqr[2];
#pragma unroll
  for (int kc = 0; kc < 2; ++kc) {
    aq[kc]  = load_a_bf16(Qp,  DH, i_base, kc * 32, lane);
    aqr[kc] = load_a_bf16(Qrp, DH, i_base, kc * 32, lane);
  }

  float rm[8], rl[8];
  v8f o[4];
#pragma unroll
  for (int v = 0; v < 8; ++v) { rm[v] = -3.0e38f; rl[v] = 0.0f; }
#pragma unroll
  for (int f = 0; f < 4; ++f) o[f] = v8f_zero();

  const int Mh = ((lane >> 4) & 1) * 8;

  for (int j0 = 0; j0 <= i_base + 15; j0 += 64) {
    if (j0 + 64 < QLEN) {                      // prefetch next tile (global_prefetch_b8)
      __builtin_prefetch(Kp + (size_t)(j0 + 64) * DH, 0, 1);
      __builtin_prefetch(Vtp + (size_t)(lane & 15) * QLEN + j0 + 64, 0, 1);
    }

    // ---- AC = (q + r_w_bias) . K^T : 4 N-subtiles x 2 K-chunks ----
    v8f s[4];
#pragma unroll
    for (int f = 0; f < 4; ++f) {
      v8f c = v8f_zero();
#pragma unroll
      for (int kc = 0; kc < 2; ++kc) {
        v16bf bk = load_b_bf16(Kp, DH, j0 + f * 16, kc * 32, lane);
        c = wmma_bf16(aqr[kc], bk, c);
      }
      s[f] = c;
    }

    // ---- banded T = q . r_emb^T over m in [m0, m0+79] -> LDS ----
    const int m0 = 2047 + j0 - i_base - 15;    // >= 0 always
#pragma unroll
    for (int sub = 0; sub < 5; ++sub) {
      v8f t = v8f_zero();
#pragma unroll
      for (int kc = 0; kc < 2; ++kc) {
        v16bf br = load_b_bf16_guard(Rp, DM, m0 + sub * 16, kc * 32, lane, QLEN);
        t = wmma_bf16(aq[kc], br, t);
      }
#pragma unroll
      for (int v = 0; v < 8; ++v)
        T_lds[wv][v + Mh][sub * 16 + (lane & 15)] = t[v];
    }

    // ---- S = (AC + BD) * SCALE with causal mask (shifted band gather) ----
#pragma unroll
    for (int f = 0; f < 4; ++f) {
#pragma unroll
      for (int v = 0; v < 8; ++v) {
        const int M  = v + Mh;
        const int jj = f * 16 + (lane & 15);
        const int jg = j0 + jj;
        const int ig = i_base + M;
        float sv;
        if (jg > ig) {
          sv = NEG_INF_F;
        } else {
          const int c = jj - M + 15;                         // 0..78
          const float bd = T_lds[wv][M][c] +
                           r_bias[(size_t)(m0 + c) * NH + n];
          sv = (s[f][v] + bd) * ATT_SCALE;
        }
        s[f][v] = sv;
      }
    }

    // ---- online softmax (each row lives in a 16-lane half-wave) ----
#pragma unroll
    for (int v = 0; v < 8; ++v) {
      float mx = rm[v];
#pragma unroll
      for (int f = 0; f < 4; ++f) mx = fmaxf(mx, s[f][v]);
#pragma unroll
      for (int off = 1; off < 16; off <<= 1) mx = fmaxf(mx, __shfl_xor(mx, off, 16));
      const float alpha = __expf(rm[v] - mx);
      rm[v] = mx;
      float ps = 0.0f;
#pragma unroll
      for (int f = 0; f < 4; ++f) {
        const float p = __expf(s[f][v] - mx);
        s[f][v] = p;
        ps += p;
      }
#pragma unroll
      for (int off = 1; off < 16; off <<= 1) ps += __shfl_xor(ps, off, 16);
      rl[v] = rl[v] * alpha + ps;
#pragma unroll
      for (int f = 0; f < 4; ++f) o[f][v] *= alpha;
    }

    // ---- P -> LDS (bf16, A-layout relayout), then O += P . V ----
#pragma unroll
    for (int f = 0; f < 4; ++f)
#pragma unroll
      for (int v = 0; v < 8; ++v)
        P_lds[wv][v + Mh][f * 16 + (lane & 15)] = f2bf(s[f][v]);

#pragma unroll
    for (int kc = 0; kc < 2; ++kc) {
      v16bf ap = load_a_bf16(&P_lds[wv][0][0], 64, 0, kc * 32, lane);
#pragma unroll
      for (int f = 0; f < 4; ++f) {
        v16bf bvv = load_b_bf16(Vtp, QLEN, f * 16, j0 + kc * 32, lane);
        o[f] = wmma_bf16(ap, bvv, o[f]);
      }
    }
  }

  // ---- normalize and store attn_vec bf16 (row = i*BSZ+b, col = n*64+d) ----
#pragma unroll
  for (int f = 0; f < 4; ++f) {
#pragma unroll
    for (int v = 0; v < 8; ++v) {
      const int M = v + Mh;
      const int i = i_base + M;
      const int d = f * 16 + (lane & 15);
      av_bf[((size_t)i * BSZ + b) * DM + n * DH + d] = f2bf(o[f][v] / rl[v]);
    }
  }
}

// ===========================================================================
// Kernel 3: O projection + residual.  av(4096x768,bf16) * o_w^T(768x768,bf16) + w.
// ===========================================================================
__global__ __launch_bounds__(256) void oproj_kernel(
    const unsigned short* __restrict__ av_bf, const unsigned short* __restrict__ ow_bf,
    const float* __restrict__ w_in, float* __restrict__ y) {
  const int lane = threadIdx.x & 31;
  const int wave = blockIdx.x * (blockDim.x >> 5) + (threadIdx.x >> 5);
  const int NCT  = DM / 64;                    // 12
  const int rs   = wave / NCT;
  const int ct   = wave % NCT;
  const int row0 = rs * 16, col0 = ct * 64;

  v8f acc[4];
#pragma unroll
  for (int f = 0; f < 4; ++f) acc[f] = v8f_zero();

  for (int k0 = 0; k0 < DM; k0 += 32) {
    v16bf a = load_a_bf16(av_bf, DM, row0, k0, lane);
#pragma unroll
    for (int f = 0; f < 4; ++f) {
      v16bf b = load_b_bf16(ow_bf, DM, col0 + f * 16, k0, lane);  // B[h][m]=o_w[m][h]
      acc[f] = wmma_bf16(a, b, acc[f]);
    }
  }

  const int Mh = ((lane >> 4) & 1) * 8;
#pragma unroll
  for (int f = 0; f < 4; ++f) {
#pragma unroll
    for (int v = 0; v < 8; ++v) {
      const int grow = row0 + v + Mh;
      const int gcol = col0 + f * 16 + (lane & 15);
      y[(size_t)grow * DM + gcol] = acc[f][v] + w_in[(size_t)grow * DM + gcol];
    }
  }
}

// ===========================================================================
// Kernel 4: in-place LayerNorm over 768 cols (one block per row).
// ===========================================================================
__global__ __launch_bounds__(256) void ln_kernel(
    float* __restrict__ y, const float* __restrict__ gamma,
    const float* __restrict__ beta) {
  __shared__ float s1[256], s2[256];
  const int r = blockIdx.x;
  const int t = threadIdx.x;
  float x[3];
  float sum = 0.0f, sq = 0.0f;
#pragma unroll
  for (int k = 0; k < 3; ++k) {
    x[k] = y[(size_t)r * DM + t + k * 256];
    sum += x[k];
    sq  += x[k] * x[k];
  }
  s1[t] = sum; s2[t] = sq;
  __syncthreads();
  for (int off = 128; off > 0; off >>= 1) {
    if (t < off) { s1[t] += s1[t + off]; s2[t] += s2[t + off]; }
    __syncthreads();
  }
  const float mu  = s1[0] * (1.0f / DM);
  const float var = s2[0] * (1.0f / DM) - mu * mu;
  const float inv = rsqrtf(var + 1e-5f);
#pragma unroll
  for (int k = 0; k < 3; ++k) {
    const int c = t + k * 256;
    y[(size_t)r * DM + c] = (x[k] - mu) * inv * gamma[c] + beta[c];
  }
}

// ===========================================================================
extern "C" void kernel_launch(void* const* d_in, const int* in_sizes, int n_in,
                              void* d_out, int out_size, void* d_ws, size_t ws_size,
                              hipStream_t stream) {
  const float* w        = (const float*)d_in[0];   // (2048, 2, 768)
  const float* r_emb    = (const float*)d_in[1];   // (2048, 12, 64)
  const float* r_w_bias = (const float*)d_in[2];   // (12, 64)
  const float* r_bias   = (const float*)d_in[3];   // (2048, 12)
  const float* qkv_w    = (const float*)d_in[4];   // (2304, 768)
  const float* o_w      = (const float*)d_in[5];   // (768, 768)
  const float* ln_gamma = (const float*)d_in[6];   // (768,)
  const float* ln_beta  = (const float*)d_in[7];   // (768,)
  // d_in[8] attn_mask: causal mask is hardcoded.
  float* out = (float*)d_out;                      // (2048, 2, 768) fp32

  // workspace carve-up (all bf16, ~46 MB total)
  const size_t HEADS_ELEMS = (size_t)BSZ * NH * QLEN * DH;   // 3,145,728
  const size_t W_ELEMS     = (size_t)QLEN * BSZ * DM;        // 3,145,728
  const size_t QKVW_ELEMS  = (size_t)3 * DM * DM;            // 1,769,472
  const size_t OW_ELEMS    = (size_t)DM * DM;                //   589,824
  const size_t REMB_ELEMS  = (size_t)QLEN * NH * DH;         // 1,572,864

  unsigned short* q_bf    = (unsigned short*)d_ws;
  unsigned short* qrw_bf  = q_bf    + HEADS_ELEMS;
  unsigned short* k_bf    = qrw_bf  + HEADS_ELEMS;
  unsigned short* vt_bf   = k_bf    + HEADS_ELEMS;
  unsigned short* av_bf   = vt_bf   + HEADS_ELEMS;
  unsigned short* w_bf    = av_bf   + W_ELEMS;
  unsigned short* qkvw_bf = w_bf    + W_ELEMS;
  unsigned short* ow_bf   = qkvw_bf + QKVW_ELEMS;
  unsigned short* remb_bf = ow_bf   + OW_ELEMS;

  // 0) one-time fp32 -> bf16 conversions (x4-vectorized elementwise)
  cvt_kernel<<<dim3((unsigned)((W_ELEMS / 4 + 255) / 256)), dim3(256), 0, stream>>>(
      w, w_bf, (int)(W_ELEMS / 4));
  cvt_kernel<<<dim3((unsigned)((QKVW_ELEMS / 4 + 255) / 256)), dim3(256), 0, stream>>>(
      qkv_w, qkvw_bf, (int)(QKVW_ELEMS / 4));
  cvt_kernel<<<dim3((unsigned)((OW_ELEMS / 4 + 255) / 256)), dim3(256), 0, stream>>>(
      o_w, ow_bf, (int)(OW_ELEMS / 4));
  cvt_kernel<<<dim3((unsigned)((REMB_ELEMS / 4 + 255) / 256)), dim3(256), 0, stream>>>(
      r_emb, remb_bf, (int)(REMB_ELEMS / 4));

  // 1) QKV projection: 256 row-strips x 36 col-tiles = 9216 wave-tiles / 8 waves
  qkv_kernel<<<dim3(1152), dim3(256), 0, stream>>>(w_bf, qkvw_bf, r_w_bias,
                                                   q_bf, qrw_bf, k_bf, vt_bf);
  // 2) flash attention: 24 (b,n) x 32 i-tiles, 4 waves each
  attn_kernel<<<dim3(768), dim3(128), 0, stream>>>(q_bf, qrw_bf, k_bf, vt_bf,
                                                   remb_bf, r_bias, av_bf);
  // 3) O projection + residual -> d_out (pre-LN)
  oproj_kernel<<<dim3(384), dim3(256), 0, stream>>>(av_bf, ow_bf, w, out);
  // 4) LayerNorm in place on d_out (block-local rows -> in-place is safe)
  ln_kernel<<<dim3(4096), dim3(256), 0, stream>>>(out, ln_gamma, ln_beta);
}
